// PinderMPNNModel_18425409700024
// MI455X (gfx1250) — matrix operand order repaired and checked by reading, not compiled, not run on hardware
//
#include <hip/hip_runtime.h>

// ============================================================================
// PinderMPNN for MI455X (gfx1250, wave32, WMMA).
// All GEMMs (D=64 feature dim) use v_wmma_f32_16x16x32_f16 with fp32 accum.
// Activations/weights stored f16 (memory-bound workload @ 23.3 TB/s HBM);
// BN statistics + affine folding done in fp32. Edge-concat GEMM fuses the
// gather of h[dst], h[src] and the pairwise distance into A-fragment build.
// ============================================================================

typedef _Float16 half_t;
typedef __attribute__((ext_vector_type(2)))  _Float16 v2h;
typedef __attribute__((ext_vector_type(8)))  _Float16 v8h;
typedef __attribute__((ext_vector_type(16))) _Float16 v16h;
typedef __attribute__((ext_vector_type(8)))  float    v8f;

static constexpr int N    = 50000;
static constexpr int E    = 800000;
static constexpr int NPAD = 50048;   // 391 * 128

__device__ inline v8f wmma16(v16h a, v16h b, v8f c) {
  // D = A(16x32 f16) x B(32x16 f16) + C(16x16 f32)
  return __builtin_amdgcn_wmma_f32_16x16x32_f16(false, a, false, b, (short)0, c,
                                                false, false);
}

// ---------------------------------------------------------------------------
// Generic GEMM: Y[rows x 64] = act(A[rows x Kpad]) * W^T(64 x Kpad) (+bias)
// act = optional per-K affine (BN fold) + ReLU applied while loading A.
// Optional per-column stats (sum, sumsq) accumulated for the *output*.
// Block: 256 threads = 8 waves; block tile = 128 rows x 64 cols,
// wave tile = 16 rows x 64 cols (4 WMMA accumulators).
// ---------------------------------------------------------------------------
__launch_bounds__(256)
__global__ void k_gemm(const half_t* __restrict__ A, int rows, int Kpad,
                       const half_t* __restrict__ W,       // 64 x Kpad f16
                       const float* __restrict__ affineA,  // [scale64|shift64] or null (Kpad==64 only)
                       const float* __restrict__ bias,     // 64 f32 or null
                       half_t* __restrict__ Y,             // rows x 64 f16
                       float* __restrict__ stats)          // 128 f32 or null
{
  extern __shared__ char smem[];
  half_t* lw = (half_t*)smem;                          // 64*Kpad halves
  float*  ls = (float*)(smem + (size_t)64 * Kpad * 2); // 128 floats

  int tid = threadIdx.x;
  {
    const unsigned* Wg = (const unsigned*)W;
    unsigned* Wl = (unsigned*)lw;
    int nwords = 64 * Kpad / 2;
    for (int i = tid; i < nwords; i += 256) Wl[i] = Wg[i];
  }
  if (stats && tid < 128) ls[tid] = 0.f;
  __syncthreads();

  int lane = tid & 31, wave = tid >> 5;
  int hi = lane >> 4, lm = lane & 15;
  long rowbase = (long)blockIdx.x * 128 + wave * 16;
  long arow    = rowbase + lm;                 // padded allocation: always readable
  const half_t* ap = A + (size_t)arow * Kpad;

  v8f acc[4] = {};
  for (int k0 = 0; k0 < Kpad; k0 += 32) {
    // A fragment: elems 0..15 -> K = k0 + hi*16 + i
    v8h lo  = *(const v8h*)(ap + k0 + hi * 16);
    v8h hi8 = *(const v8h*)(ap + k0 + hi * 16 + 8);
    v16h af;
    if (affineA) {
#pragma unroll
      for (int i = 0; i < 8; i++) {
        int ka = k0 + hi * 16 + i;
        int kb = k0 + hi * 16 + 8 + i;
        af[i]     = (half_t)fmaxf(affineA[ka] * (float)lo[i]  + affineA[64 + ka], 0.f);
        af[i + 8] = (half_t)fmaxf(affineA[kb] * (float)hi8[i] + affineA[64 + kb], 0.f);
      }
    } else {
#pragma unroll
      for (int i = 0; i < 8; i++) { af[i] = lo[i]; af[i + 8] = hi8[i]; }
    }
#pragma unroll
    for (int ct = 0; ct < 4; ct++) {
      const v16h* bp = (const v16h*)(lw + (size_t)(ct * 16 + lm) * Kpad + k0 + hi * 16);
      acc[ct] = wmma16(af, *bp, acc[ct]);
    }
  }

#pragma unroll
  for (int ct = 0; ct < 4; ct++) {
    int col = ct * 16 + lm;
    float bs = bias ? bias[col] : 0.f;
    float ssum = 0.f, ssq = 0.f;
#pragma unroll
    for (int j = 0; j < 8; j++) {
      long row = rowbase + j + hi * 8;   // C/D layout: VGPR j -> M=j (lanes<16) / j+8
      float v = acc[ct][j] + bs;
      if (row < rows) {
        Y[(size_t)row * 64 + col] = (half_t)v;
        ssum += v; ssq += v * v;
      }
    }
    if (stats) { atomicAdd(&ls[col], ssum); atomicAdd(&ls[64 + col], ssq); }
  }
  if (stats) {
    __syncthreads();
    if (tid < 128) atomicAdd(&stats[tid], ls[tid]);
  }
}

// ---------------------------------------------------------------------------
// Edge GEMM with fused gather: A row e = [h[dst[e]](64) | h[src[e]](64) |
// dist(1) | zeros(31)]  (Kpad = 160).  Also emits pos_diff (E x 3).
// ---------------------------------------------------------------------------
__launch_bounds__(256)
__global__ void k_gemm_edge(const half_t* __restrict__ h,
                            const float* __restrict__ pos,
                            const int* __restrict__ srcI,
                            const int* __restrict__ dstI,
                            const half_t* __restrict__ W,    // 64 x 160 (padded)
                            int rows,
                            half_t* __restrict__ Y,          // E x 64
                            float* __restrict__ pdiff,       // E x 3
                            float* __restrict__ stats)       // 128
{
  __shared__ half_t lw[64 * 160];
  __shared__ float  ls[128];
  int tid = threadIdx.x;
  {
    const unsigned* Wg = (const unsigned*)W;
    unsigned* Wl = (unsigned*)lw;
    for (int i = tid; i < 64 * 160 / 2; i += 256) Wl[i] = Wg[i];
  }
  if (tid < 128) ls[tid] = 0.f;
  __syncthreads();

  int lane = tid & 31, wave = tid >> 5;
  int hi = lane >> 4, lm = lane & 15;
  long rowbase = (long)blockIdx.x * 128 + wave * 16;
  long e = rowbase + lm;                       // E is a multiple of 128
  int id = dstI[e], is = srcI[e];
  const half_t* hd = h + (size_t)id * 64;      // h_i = h[dst]
  const half_t* hs = h + (size_t)is * 64;      // h_j = h[src]
  float dx = pos[id * 3 + 0] - pos[is * 3 + 0];
  float dy = pos[id * 3 + 1] - pos[is * 3 + 1];
  float dz = pos[id * 3 + 2] - pos[is * 3 + 2];
  float dist = sqrtf(dx * dx + dy * dy + dz * dz + 1e-12f);
  if (hi == 0) {
    pdiff[e * 3 + 0] = dx; pdiff[e * 3 + 1] = dy; pdiff[e * 3 + 2] = dz;
  }

  v8f acc[4] = {};
#pragma unroll
  for (int kc = 0; kc < 5; kc++) {
    v16h af;
    if (kc < 4) {
      const half_t* base = (kc < 2) ? (hd + kc * 32) : (hs + (kc - 2) * 32);
      v8h lo  = *(const v8h*)(base + hi * 16);
      v8h hi8 = *(const v8h*)(base + hi * 16 + 8);
#pragma unroll
      for (int i = 0; i < 8; i++) { af[i] = lo[i]; af[i + 8] = hi8[i]; }
    } else {
#pragma unroll
      for (int i = 0; i < 16; i++) af[i] = (half_t)0.f;
      if (hi == 0) af[0] = (half_t)dist;       // K = 128 column
    }
    int k0 = kc * 32;
#pragma unroll
    for (int ct = 0; ct < 4; ct++) {
      const v16h* bp = (const v16h*)(lw + (size_t)(ct * 16 + lm) * 160 + k0 + hi * 16);
      acc[ct] = wmma16(af, *bp, acc[ct]);
    }
  }

#pragma unroll
  for (int ct = 0; ct < 4; ct++) {
    int col = ct * 16 + lm;
    float ssum = 0.f, ssq = 0.f;
#pragma unroll
    for (int j = 0; j < 8; j++) {
      long row = rowbase + j + hi * 8;
      float v = acc[ct][j];
      if (row < rows) {
        Y[(size_t)row * 64 + col] = (half_t)v;
        ssum += v; ssq += v * v;
      }
    }
    atomicAdd(&ls[col], ssum); atomicAdd(&ls[64 + col], ssq);
  }
  __syncthreads();
  if (tid < 128) atomicAdd(&stats[tid], ls[tid]);
}

// ---------------------------------------------------------------------------
// Small helper kernels
// ---------------------------------------------------------------------------
__global__ void k_w_convert(const float* __restrict__ src, half_t* __restrict__ dst,
                            int outn, int Kin, int Kpad)
{
  int i = blockIdx.x * blockDim.x + threadIdx.x;
  if (i >= outn * Kpad) return;
  int o = i / Kpad, k = i % Kpad;
  dst[i] = (k < Kin) ? (half_t)src[o * Kin + k] : (half_t)0.f;
}

__global__ void k_bn_finalize(const float* __restrict__ stats, const float* __restrict__ g,
                              const float* __restrict__ be, float inv_n,
                              float* __restrict__ aff)
{
  int c = threadIdx.x; if (c >= 64) return;
  float m = stats[c] * inv_n;
  float v = stats[64 + c] * inv_n - m * m;
  float rs = rsqrtf(v + 1e-5f);
  float sc = g[c] * rs;
  aff[c] = sc;
  aff[64 + c] = be[c] - m * sc;
}

__global__ void k_bn_relu_apply(const half_t* __restrict__ yin, const float* __restrict__ aff,
                                half_t* __restrict__ yout, long count)
{
  long i = (long)blockIdx.x * blockDim.x + threadIdx.x;
  if (i >= count) return;
  int c = (int)(i & 63);
  float x = aff[c] * (float)yin[i] + aff[64 + c];
  yout[i] = (half_t)fmaxf(x, 0.f);
}

__global__ void k_count(const int* __restrict__ dstI, int e_n, float* __restrict__ cnt)
{
  int e = blockIdx.x * blockDim.x + threadIdx.x;
  if (e < e_n) atomicAdd(&cnt[dstI[e]], 1.f);
}

__global__ void k_scatter_m(const half_t* __restrict__ m, const int* __restrict__ dstI,
                            int e_n, float* __restrict__ aggr)
{
  long t = (long)blockIdx.x * blockDim.x + threadIdx.x;
  long e = t >> 5; int lane = (int)(t & 31);
  if (e >= e_n) return;
  int d = dstI[e];
  v2h val = *(const v2h*)(m + (size_t)e * 64 + lane * 2);
  atomicAdd(&aggr[(size_t)d * 64 + lane * 2 + 0], (float)val[0]);
  atomicAdd(&aggr[(size_t)d * 64 + lane * 2 + 1], (float)val[1]);
}

// One wave per edge: s = pos_w2 . relu(bn(y3)) + b2 ; scatter pos_diff * s
__global__ void k_pos_scatter(const half_t* __restrict__ y3, const float* __restrict__ aff,
                              const float* __restrict__ w2, const float* __restrict__ b2,
                              const float* __restrict__ pd, const int* __restrict__ dstI,
                              int e_n, float* __restrict__ paggr)
{
  long t = (long)blockIdx.x * blockDim.x + threadIdx.x;
  long e = t >> 5; int lane = (int)(t & 31);
  if (e >= e_n) return;
  float p = 0.f;
  for (int i = lane; i < 64; i += 32) {
    float x = fmaxf(aff[i] * (float)y3[(size_t)e * 64 + i] + aff[64 + i], 0.f);
    p += x * w2[i];
  }
  for (int off = 16; off > 0; off >>= 1) p += __shfl_xor(p, off);
  float s = p + b2[0];
  if (lane < 3) atomicAdd(&paggr[(size_t)dstI[e] * 3 + lane], pd[e * 3 + lane] * s);
}

__global__ void k_lin_in(const float* __restrict__ x, const float* __restrict__ w,
                         const float* __restrict__ b, int n, half_t* __restrict__ hout)
{
  long i = (long)blockIdx.x * blockDim.x + threadIdx.x;
  if (i >= (long)n * 64) return;
  long r = i >> 6; int c = (int)(i & 63);
  hout[i] = (half_t)(x[r] * w[c] + b[c]);
}

__global__ void k_build_u(const half_t* __restrict__ h, const float* __restrict__ maggr,
                          const float* __restrict__ cnt, int n, half_t* __restrict__ u)
{
  long i = (long)blockIdx.x * blockDim.x + threadIdx.x;
  if (i >= (long)n * 64) return;
  long r = i >> 6; int c = (int)(i & 63);
  u[r * 128 + c] = h[i];
  float ct = fmaxf(cnt[r], 1.f);
  u[r * 128 + 64 + c] = (half_t)(maggr[i] / ct);
}

__global__ void k_pos_update(float* __restrict__ p, const float* __restrict__ paggr,
                             const float* __restrict__ cnt, int n)
{
  long i = (long)blockIdx.x * blockDim.x + threadIdx.x;
  if (i >= (long)n * 3) return;
  long r = i / 3;
  p[i] += paggr[i] / fmaxf(cnt[r], 1.f);
}

__global__ void k_colsum(const half_t* __restrict__ h, int n, float* __restrict__ out)
{
  __shared__ float ls[256];
  int tid = threadIdx.x;
  int col = tid & 63, rl = tid >> 6;
  long base = (long)blockIdx.x * 256;
  float s = 0.f;
  for (int i = 0; i < 64; i++) {
    long r = base + i * 4 + rl;
    if (r < n) s += (float)h[(size_t)r * 64 + col];
  }
  ls[tid] = s;
  __syncthreads();
  if (rl == 0) atomicAdd(&out[col], ls[col] + ls[64 + col] + ls[128 + col] + ls[192 + col]);
}

// seq-len-1 MHA degenerates: attn weight == 1, so a = out_w @ (W_v kv + b_v) + out_b
__global__ void k_heads(const float* __restrict__ hsr, const float* __restrict__ hsl, float invN,
    const float* riw, const float* rib, const float* row_, const float* rob,
    const float* liw, const float* lib, const float* low_, const float* lob,
    const float* rrw, const float* rrb, const float* rlw, const float* rlb,
    const float* trw, const float* trb, const float* tlw, const float* tlb,
    float* __restrict__ Rt)
{
  __shared__ float hr[64], hl[64], v[64], ar[64], al[64];
  int t = threadIdx.x;
  hr[t] = hsr[t] * invN; hl[t] = hsl[t] * invN;
  __syncthreads();
  float acc = rib[128 + t];
  for (int k = 0; k < 64; k++) acc += riw[(128 + t) * 64 + k] * hl[k];
  v[t] = acc; __syncthreads();
  acc = rob[t];
  for (int k = 0; k < 64; k++) acc += row_[t * 64 + k] * v[k];
  ar[t] = acc; __syncthreads();
  acc = lib[128 + t];
  for (int k = 0; k < 64; k++) acc += liw[(128 + t) * 64 + k] * hr[k];
  v[t] = acc; __syncthreads();
  acc = lob[t];
  for (int k = 0; k < 64; k++) acc += low_[t * 64 + k] * v[k];
  al[t] = acc; __syncthreads();
  if (t < 9) {
    float r = rrb[t];  for (int k = 0; k < 64; k++) r += rrw[t * 64 + k] * ar[k]; Rt[t] = r;
    float r2 = rlb[t]; for (int k = 0; k < 64; k++) r2 += rlw[t * 64 + k] * al[k]; Rt[12 + t] = r2;
  }
  if (t < 3) {
    float r = trb[t];  for (int k = 0; k < 64; k++) r += trw[t * 64 + k] * ar[k]; Rt[9 + t] = r;
    float r2 = tlb[t]; for (int k = 0; k < 64; k++) r2 += tlw[t * 64 + k] * al[k]; Rt[21 + t] = r2;
  }
}

__global__ void k_transform(const float* __restrict__ pos, const float* __restrict__ Rt,
                            int n, float* __restrict__ out)
{
  long i = (long)blockIdx.x * blockDim.x + threadIdx.x;
  if (i >= n) return;
  float x = pos[i * 3], y = pos[i * 3 + 1], z = pos[i * 3 + 2];
  const float* R = Rt; const float* tr = Rt + 9;
  out[i * 3 + 0] = R[0] * x + R[1] * y + R[2] * z + tr[0];
  out[i * 3 + 1] = R[3] * x + R[4] * y + R[5] * z + tr[1];
  out[i * 3 + 2] = R[6] * x + R[7] * y + R[8] * z + tr[2];
}

// ---------------------------------------------------------------------------
// Host-side orchestration.
// Input order: 6 top-level arrays, then jax tree-leaves of `params`
// (dicts flatten with sorted keys, lists in index order).
// ---------------------------------------------------------------------------
static constexpr int IN_XR = 0, IN_PR = 1, IN_EIR = 2, IN_XL = 3, IN_PL = 4, IN_EIL = 5;
static constexpr int P0 = 6;
// layer leaves, sorted keys:
enum { L_MSG_B1 = 0, L_MSG_B2, L_MSG_BE1, L_MSG_BE2, L_MSG_G1, L_MSG_G2, L_MSG_W1, L_MSG_W2,
       L_OUT_B, L_OUT_W, L_POS_B1, L_POS_B2, L_POS_BE1, L_POS_G1, L_POS_W1, L_POS_W2,
       L_UPD_B1, L_UPD_B2, L_UPD_BE1, L_UPD_BE2, L_UPD_G1, L_UPD_G2, L_UPD_W1, L_UPD_W2 };
// sorted top-level keys of params:
static constexpr int LIG_ATTN = P0;            // in_b,in_w,out_b,out_w
static constexpr int LIG_LAY0 = P0 + 4;
static constexpr int LIG_LAY1 = P0 + 28;
static constexpr int LIN_LIG_B = P0 + 52, LIN_LIG_W = P0 + 53;
static constexpr int LIN_REC_B = P0 + 54, LIN_REC_W = P0 + 55;
static constexpr int REC_ATTN = P0 + 56;
static constexpr int REC_LAY0 = P0 + 60;
static constexpr int REC_LAY1 = P0 + 84;
static constexpr int ROT_LIG_B = P0 + 108, ROT_LIG_W = P0 + 109;
static constexpr int ROT_REC_B = P0 + 110, ROT_REC_W = P0 + 111;
static constexpr int TR_LIG_B = P0 + 112, TR_LIG_W = P0 + 113;
static constexpr int TR_REC_B = P0 + 114, TR_REC_W = P0 + 115;

static inline unsigned cdivu(long a, long b) { return (unsigned)((a + b - 1) / b); }

extern "C" void kernel_launch(void* const* d_in, const int* in_sizes, int n_in,
                              void* d_out, int out_size, void* d_ws, size_t ws_size,
                              hipStream_t stream)
{
  (void)in_sizes; (void)n_in; (void)out_size; (void)ws_size;
  char* ws = (char*)d_ws;
  size_t off = 0;
  auto alloc = [&](size_t bytes) -> char* {
    char* p = ws + off;
    off = (off + bytes + 255) & ~(size_t)255;
    return p;
  };

  half_t *wf_msg1[4], *wf_msg2[4], *wf_pos1[4], *wf_upd1[4], *wf_upd2[4], *wf_out[4];
  for (int s = 0; s < 4; s++) {
    wf_msg1[s] = (half_t*)alloc(64 * 160 * 2);
    wf_msg2[s] = (half_t*)alloc(64 * 64 * 2);
    wf_pos1[s] = (half_t*)alloc(64 * 64 * 2);
    wf_upd1[s] = (half_t*)alloc(64 * 128 * 2);
    wf_upd2[s] = (half_t*)alloc(64 * 64 * 2);
    wf_out[s]  = (half_t*)alloc(64 * 64 * 2);
  }
  float* statsA = (float*)alloc(20 * 128 * 4);
  float* affA   = (float*)alloc(20 * 128 * 4);
  float* hsum   = (float*)alloc(2 * 64 * 4);
  float* Rt     = (float*)alloc(24 * 4);
  half_t* hA  = (half_t*)alloc((size_t)NPAD * 64 * 2);
  half_t* hB  = (half_t*)alloc((size_t)NPAD * 64 * 2);
  half_t* U   = (half_t*)alloc((size_t)NPAD * 128 * 2);
  half_t* Yn1 = (half_t*)alloc((size_t)NPAD * 64 * 2);
  half_t* Yn2 = (half_t*)alloc((size_t)NPAD * 64 * 2);
  float* Ppos = (float*)alloc((size_t)N * 3 * 4);
  size_t aggrBytes = (size_t)N * 64 * 4 + (size_t)N * 3 * 4 + (size_t)N * 4;
  char* aggr = alloc(aggrBytes);
  float* m_aggr = (float*)aggr;
  float* p_aggr = (float*)(aggr + (size_t)N * 64 * 4);
  float* cnt    = (float*)(aggr + (size_t)N * 64 * 4 + (size_t)N * 3 * 4);
  half_t* B1 = (half_t*)alloc((size_t)E * 64 * 2);
  half_t* B2 = (half_t*)alloc((size_t)E * 64 * 2);
  float*  Pd = (float*)alloc((size_t)E * 3 * 4);

  auto F = [&](int i) -> const float* { return (const float*)d_in[i]; };
  auto layer_base = [&](int g, int l) -> int {
    return (g == 0) ? (l == 0 ? REC_LAY0 : REC_LAY1) : (l == 0 ? LIG_LAY0 : LIG_LAY1);
  };

  // --- weight conversion to f16 (K padded to multiple of 32) ---
  for (int s = 0; s < 4; s++) {
    int lb = layer_base(s >> 1, s & 1);
    k_w_convert<<<cdivu(64 * 160, 256), 256, 0, stream>>>(F(lb + L_MSG_W1), wf_msg1[s], 64, 129, 160);
    k_w_convert<<<cdivu(64 * 64, 256), 256, 0, stream>>>(F(lb + L_MSG_W2), wf_msg2[s], 64, 64, 64);
    k_w_convert<<<cdivu(64 * 64, 256), 256, 0, stream>>>(F(lb + L_POS_W1), wf_pos1[s], 64, 64, 64);
    k_w_convert<<<cdivu(64 * 128, 256), 256, 0, stream>>>(F(lb + L_UPD_W1), wf_upd1[s], 64, 128, 128);
    k_w_convert<<<cdivu(64 * 64, 256), 256, 0, stream>>>(F(lb + L_UPD_W2), wf_upd2[s], 64, 64, 64);
    k_w_convert<<<cdivu(64 * 64, 256), 256, 0, stream>>>(F(lb + L_OUT_W), wf_out[s], 64, 64, 64);
  }
  hipMemsetAsync(hsum, 0, 2 * 64 * 4, stream);

  const size_t smem64  = (size_t)64 * 64 * 2 + 512;
  const size_t smem128 = (size_t)64 * 128 * 2 + 512;

  for (int g = 0; g < 2; ++g) {
    const float* x   = F(g == 0 ? IN_XR : IN_XL);
    const float* pos = F(g == 0 ? IN_PR : IN_PL);
    const int* ei    = (const int*)d_in[g == 0 ? IN_EIR : IN_EIL];
    const int* srcI = ei;          // edge_index[0]
    const int* dstI = ei + E;      // edge_index[1]

    k_lin_in<<<cdivu((long)N * 64, 256), 256, 0, stream>>>(
        x, F(g == 0 ? LIN_REC_W : LIN_LIG_W), F(g == 0 ? LIN_REC_B : LIN_LIG_B), N, hA);
    hipMemcpyAsync(Ppos, pos, (size_t)N * 3 * 4, hipMemcpyDeviceToDevice, stream);

    half_t* hcur = hA; half_t* hnxt = hB;
    for (int l = 0; l < 2; ++l) {
      int s = g * 2 + l;
      int lb = layer_base(g, l);
      float* st = statsA + (size_t)s * 5 * 128;
      float* af = affA   + (size_t)s * 5 * 128;
      hipMemsetAsync(st, 0, 5 * 128 * 4, stream);
      hipMemsetAsync(aggr, 0, aggrBytes, stream);
      k_count<<<cdivu(E, 256), 256, 0, stream>>>(dstI, E, cnt);

      // msg MLP stage 1 (fused gather + K=160 GEMM) -> y1 + stats
      k_gemm_edge<<<E / 128, 256, 0, stream>>>(hcur, Ppos, srcI, dstI, wf_msg1[s], E, B1, Pd, st);
      k_bn_finalize<<<1, 64, 0, stream>>>(st, F(lb + L_MSG_G1), F(lb + L_MSG_BE1), 1.f / E, af);
      // msg MLP stage 2: relu(bn(y1)) folded into A-load
      k_gemm<<<E / 128, 256, smem64, stream>>>(B1, E, 64, wf_msg2[s], af, nullptr, B2, st + 128);
      k_bn_finalize<<<1, 64, 0, stream>>>(st + 128, F(lb + L_MSG_G2), F(lb + L_MSG_BE2), 1.f / E, af + 128);
      // materialize m_act = relu(bn(y2)) (needed by aggregation AND pos MLP)
      k_bn_relu_apply<<<cdivu((long)E * 64, 256), 256, 0, stream>>>(B2, af + 128, B1, (long)E * 64);
      // pos MLP stage 1 -> y3 + stats
      k_gemm<<<E / 128, 256, smem64, stream>>>(B1, E, 64, wf_pos1[s], nullptr, nullptr, B2, st + 256);
      k_bn_finalize<<<1, 64, 0, stream>>>(st + 256, F(lb + L_POS_G1), F(lb + L_POS_BE1), 1.f / E, af + 256);
      // aggregations (segment sums via fp32 atomics)
      k_scatter_m<<<cdivu((long)E * 32, 256), 256, 0, stream>>>(B1, dstI, E, m_aggr);
      k_pos_scatter<<<cdivu((long)E * 32, 256), 256, 0, stream>>>(
          B2, af + 256, F(lb + L_POS_W2), F(lb + L_POS_B2), Pd, dstI, E, p_aggr);
      // update MLP (node phase)
      k_build_u<<<cdivu((long)N * 64, 256), 256, 0, stream>>>(hcur, m_aggr, cnt, N, U);
      k_gemm<<<NPAD / 128, 256, smem128, stream>>>(U, N, 128, wf_upd1[s], nullptr, nullptr, Yn1, st + 384);
      k_bn_finalize<<<1, 64, 0, stream>>>(st + 384, F(lb + L_UPD_G1), F(lb + L_UPD_BE1), 1.f / N, af + 384);
      k_gemm<<<NPAD / 128, 256, smem64, stream>>>(Yn1, N, 64, wf_upd2[s], af + 384, nullptr, Yn2, st + 512);
      k_bn_finalize<<<1, 64, 0, stream>>>(st + 512, F(lb + L_UPD_G2), F(lb + L_UPD_BE2), 1.f / N, af + 512);
      k_gemm<<<NPAD / 128, 256, smem64, stream>>>(Yn2, N, 64, wf_out[s], af + 512, F(lb + L_OUT_B), hnxt, nullptr);
      k_pos_update<<<cdivu((long)N * 3, 256), 256, 0, stream>>>(Ppos, p_aggr, cnt, N);
      half_t* t = hcur; hcur = hnxt; hnxt = t;
    }
    k_colsum<<<cdivu(N, 256), 256, 0, stream>>>(hcur, N, hsum + g * 64);
  }

  k_heads<<<1, 64, 0, stream>>>(hsum, hsum + 64, 1.f / N,
      F(REC_ATTN + 1), F(REC_ATTN + 0), F(REC_ATTN + 3), F(REC_ATTN + 2),
      F(LIG_ATTN + 1), F(LIG_ATTN + 0), F(LIG_ATTN + 3), F(LIG_ATTN + 2),
      F(ROT_REC_W), F(ROT_REC_B), F(ROT_LIG_W), F(ROT_LIG_B),
      F(TR_REC_W), F(TR_REC_B), F(TR_LIG_W), F(TR_LIG_B), Rt);

  float* out = (float*)d_out;
  k_transform<<<cdivu(N, 256), 256, 0, stream>>>(F(IN_PR), Rt, N, out);
  k_transform<<<cdivu(N, 256), 256, 0, stream>>>(F(IN_PL), Rt + 12, N, out + (size_t)N * 3);
}
